// NeuralODE_54030688583744
// MI455X (gfx1250) — matrix operand order repaired and checked by reading, not compiled, not run on hardware
//
#include <hip/hip_runtime.h>

typedef __attribute__((ext_vector_type(16))) _Float16 v16h;
typedef __attribute__((ext_vector_type(8)))  _Float16 v8h;
typedef __attribute__((ext_vector_type(4)))  _Float16 v4h;
typedef __attribute__((ext_vector_type(8)))  float    v8f;
typedef __attribute__((ext_vector_type(4)))  float    v4f;
typedef __attribute__((ext_vector_type(4)))  int      v4i;

#define B_N   2048
#define T_N   200
#define D_N   64
#define W_N   256
#define MT    16          // batch rows per workgroup
#define NWAVE 8           // 256 threads = 8 wave32
#define HA_S  264         // padded halves per row of A-staging (breaks LDS bank conflicts)

// fragment blob layout in d_ws, in halves (WMMA B-fragment order, lane-major)
#define W1F_OFF 0
#define W1F_LEN (16 * 2 * 512)   // 16 n-tiles, K=64 -> 2 ksteps
#define W2F_OFF (W1F_OFF + W1F_LEN)
#define W2F_LEN (16 * 8 * 512)   // 16 n-tiles, K=256 -> 8 ksteps
#define W3F_OFF (W2F_OFF + W2F_LEN)
#define W3F_LEN (16 * 8 * 512)
#define W4F_OFF (W3F_OFF + W3F_LEN)
#define W4F_LEN (4 * 8 * 512)    // 4 n-tiles (N=64), K=256
#define FRAG_TOTAL (W4F_OFF + W4F_LEN)   // 163840 halves = 320 KB

// ---- Tsit5 coefficients ----
constexpr float AC[6][5] = {
  {0.f, 0.f, 0.f, 0.f, 0.f},
  {0.161f, 0.f, 0.f, 0.f, 0.f},
  {-0.008480655492356989f, 0.335480655492357f, 0.f, 0.f, 0.f},
  {2.8971530571054935f, -6.359448489975075f, 4.3622954328695815f, 0.f, 0.f},
  {5.325864828439257f, -11.748883564062828f, 7.4955393428898365f, -0.09249506636175525f, 0.f},
  {5.86145544294642f, -12.92096931784711f, 8.159367898576159f, -0.071584973281401f, -0.028269050394068383f}};
constexpr float BW[6] = {0.09646076681806523f, 0.01f, 0.4798896504144996f,
                         1.379008574103742f, -3.290069515436081f, 2.324710524099774f};

// ---- fast activations on raw trans ops (no denorm-guard ladders) ----
__device__ __forceinline__ float fast_exp2(float x) {
#if __has_builtin(__builtin_amdgcn_exp2f)
  return __builtin_amdgcn_exp2f(x);
#else
  return exp2f(x);
#endif
}
__device__ __forceinline__ float fast_log2(float x) {   // caller guarantees x in [1,2]
#if __has_builtin(__builtin_amdgcn_logf)
  return __builtin_amdgcn_logf(x);
#else
  return __log2f(x);
#endif
}
__device__ __forceinline__ float fast_softplus(float x) {
  // softplus(x) = max(x,0) + ln(1 + exp(-|x|)); argument of log is in [1,2]
  float e = fast_exp2(-fabsf(x) * 1.4426950408889634f);
  return fmaxf(x, 0.0f) + 0.6931471805599453f * fast_log2(1.0f + e);
}
__device__ __forceinline__ float fast_tanh(float x) {
#if __has_builtin(__builtin_amdgcn_tanhf)
  return __builtin_amdgcn_tanhf(x);
#else
  float xc = fminf(fmaxf(x, -9.0f), 9.0f);          // avoid inf/inf
  float t  = fast_exp2(xc * 2.8853900817779268f);   // e^(2x)
#if __has_builtin(__builtin_amdgcn_rcpf)
  return (t - 1.0f) * __builtin_amdgcn_rcpf(t + 1.0f);
#else
  return (t - 1.0f) / (t + 1.0f);
#endif
#endif
}

// ------------------------------------------------------------------
// Prep: convert fp32 weights [N_out][K_in] into f16 WMMA-B fragments.
//   B[K][N] = W[N][K], N = tile*16 + (L&15), K = kk*32 + (L>>4)*16 + i
// stored lane-major so each lane loads its 16 halves as one 32B vector.
// ------------------------------------------------------------------
__global__ void prep_frags(const float* __restrict__ W1, const float* __restrict__ W2,
                           const float* __restrict__ W3, const float* __restrict__ W4,
                           _Float16* __restrict__ frag)
{
  int idx = blockIdx.x * blockDim.x + threadIdx.x;
  if (idx >= FRAG_TOTAL) return;
  const float* src; int Kin, base;
  if (idx < W2F_OFF)      { src = W1; Kin = 64;  base = W1F_OFF; }
  else if (idx < W3F_OFF) { src = W2; Kin = 256; base = W2F_OFF; }
  else if (idx < W4F_OFF) { src = W3; Kin = 256; base = W3F_OFF; }
  else                    { src = W4; Kin = 256; base = W4F_OFF; }
  int local = idx - base;
  int i    = local & 15;
  int lane = (local >> 4) & 31;
  int fidx = local >> 9;
  int ksteps = Kin >> 5;
  int kk   = fidx % ksteps;
  int tile = fidx / ksteps;
  int N = tile * 16 + (lane & 15);
  int K = kk * 32 + (lane >> 4) * 16 + i;
  frag[idx] = (_Float16)src[N * Kin + K];
}

// ------------------------------------------------------------------
// One GEMM layer.  PAIR=true: wave handles tiles {wave, wave+8} with a
// shared A-fragment per k-step (two independent WMMA chains).
// PAIR=false (N=64 layer): waves 0-3 handle one tile each.
// ------------------------------------------------------------------
template<int KSTEPS, bool B_LDS, bool SOFTPLUS, bool PAIR>
__device__ __forceinline__ void gemm_layer(
    int wave, int lane,
    const _Float16* Ain,                       // LDS, padded rows (HA_S)
    const _Float16* Blds, const _Float16* Bglob,
    const float* bias,                         // LDS
    _Float16* hOut,                            // LDS (softplus path)
    float* kOut,                               // LDS (tanh path)
    float oscale)
{
  if (!PAIR && wave >= 4) return;              // caller still barriers after
  const int arow  = lane & 15;
  const int akb   = (lane >> 4) * 16;          // A-fragment K base (contiguous layout)
  const int rbase = (lane >> 4) * 8;           // C/D: lanes 16-31 hold M = 8..15
  const int ncol  = lane & 15;
  const _Float16* Bbase = B_LDS ? Blds : Bglob;

  v8f acc0 = {}, acc1 = {};
  #pragma unroll
  for (int kk = 0; kk < KSTEPS; ++kk) {
    const _Float16* pa = Ain + arow * HA_S + kk * 32 + akb;
    v8h a0 = *(const v8h*)pa;                  // K = base .. base+7
    v8h a1 = *(const v8h*)(pa + 8);            // K = base+8 .. base+15
    v16h A = __builtin_shufflevector(a0, a1,
               0,1,2,3,4,5,6,7,8,9,10,11,12,13,14,15);
    v16h B0 = *(const v16h*)(Bbase + ((size_t)(wave * KSTEPS + kk) * 32 + lane) * 16);
    acc0 = __builtin_amdgcn_wmma_f32_16x16x32_f16(
             false, A, false, B0, (short)0, acc0, false, false);
    if (PAIR) {
      v16h B1 = *(const v16h*)(Bbase + ((size_t)((wave + 8) * KSTEPS + kk) * 32 + lane) * 16);
      acc1 = __builtin_amdgcn_wmma_f32_16x16x32_f16(
               false, A, false, B1, (short)0, acc1, false, false);
    }
  }

  #pragma unroll
  for (int p = 0; p < (PAIR ? 2 : 1); ++p) {
    const v8f& acc = p ? acc1 : acc0;
    const int n = (wave + p * 8) * 16 + ncol;
    const float bi = bias[n];
    if (SOFTPLUS) {
      #pragma unroll
      for (int r = 0; r < 8; ++r)
        hOut[(rbase + r) * HA_S + n] = (_Float16)fast_softplus(acc[r] + bi);
    } else {
      #pragma unroll
      for (int r = 0; r < 8; ++r)
        kOut[(rbase + r) * D_N + n] = fast_tanh(acc[r] + bi) * oscale;
    }
  }
}

// ------------------------------------------------------------------
// Main: each workgroup integrates 16 batch rows through all 199 steps.
// ------------------------------------------------------------------
__global__ __launch_bounds__(256, 1) void tsit5_kernel(
    const float* __restrict__ ts, const float* __restrict__ y0,
    const float* __restrict__ b1, const float* __restrict__ b2,
    const float* __restrict__ b3, const float* __restrict__ b4,
    const float* __restrict__ oscale_p,
    const _Float16* __restrict__ frag, float* __restrict__ out)
{
  extern __shared__ v16h smem_v[];           // 32B-aligned dynamic LDS
  _Float16* sW2 = (_Float16*)smem_v;          // 65536 halves (128 KB)
  _Float16* sW3 = sW2 + W2F_LEN;              // 65536 halves (128 KB)
  _Float16* hA0 = sW3 + W3F_LEN;              // 16*264 halves
  _Float16* hA1 = hA0 + MT * HA_S;            // 16*264 halves
  float* sy  = (float*)(hA1 + MT * HA_S);     // 16*64 f32
  float* sk  = sy + MT * D_N;                 // 6 * 16*64 f32
  float* sb1 = sk + 6 * MT * D_N;             // 256
  float* sb2 = sb1 + W_N;                     // 256
  float* sb3 = sb2 + W_N;                     // 256
  float* sb4 = sb3 + W_N;                     // 64

  const int tid  = threadIdx.x;
  const int lane = tid & 31;
  const int wave = tid >> 5;
  const int row0 = blockIdx.x * MT;
  // each thread owns 4 contiguous state elements (1024 = 256*4)
  const int e_row = tid >> 4;                 // 0..15
  const int e_d   = (tid & 15) * 4;           // 0..60
  const int e_i   = e_row * D_N + e_d;        // == tid*4

  // Warm L2/WGP$ with the global-resident fragment segments (W1, W4).
  for (int i = tid * 64; i < W1F_LEN; i += 256 * 64)
    __builtin_prefetch(frag + W1F_OFF + i, 0, 2);
  for (int i = tid * 64; i < W4F_LEN; i += 256 * 64)
    __builtin_prefetch(frag + W4F_OFF + i, 0, 2);

  // Stage W2+W3 fragment blobs (contiguous in d_ws) into LDS.
#if defined(__gfx1250__) && \
    __has_builtin(__builtin_amdgcn_global_load_async_to_lds_b128) && \
    __has_builtin(__builtin_amdgcn_s_wait_asynccnt)
  {
    v4i* g = (v4i*)(frag + W2F_OFF);           // global source
    v4i* l = (v4i*)sW2;                        // LDS destination
    const int total16 = (W2F_LEN + W3F_LEN) / 8;   // # of 16-byte chunks
    for (int i = tid; i < total16; i += 256)
      __builtin_amdgcn_global_load_async_to_lds_b128(g + i, l + i, 0, 0);
    __builtin_amdgcn_s_wait_asynccnt(0);
  }
#else
  {
    const v8h* src = (const v8h*)(frag + W2F_OFF);
    v8h* dst = (v8h*)sW2;
    const int n8 = (W2F_LEN + W3F_LEN) / 8;
    for (int i = tid; i < n8; i += 256) dst[i] = src[i];
  }
#endif
  // biases (blockDim == 256 == W_N: straight-line)
  sb1[tid] = b1[tid]; sb2[tid] = b2[tid]; sb3[tid] = b3[tid];
  if (tid < D_N) sb4[tid] = b4[tid];
  // y init + t=0 snapshot (vectorized, NT store for the streamed output)
  {
    v4f y4 = *(const v4f*)(y0 + row0 * D_N + e_i);
    *(v4f*)(sy + e_i) = y4;
    __builtin_nontemporal_store(
        y4, (v4f*)(out + (size_t)(row0 + e_row) * (T_N * D_N) + e_d));
  }
  __syncthreads();

  const float oscale = oscale_p[0];

  #pragma unroll 1
  for (int t = 0; t < T_N - 1; ++t) {
    const float dt = ts[t + 1] - ts[t];

    #pragma unroll
    for (int s = 0; s < 6; ++s) {
      // Build stage input y + dt * sum_j AC[s][j] * k_j -> hA0 (f16), straight-line
      {
        v4f v = *(const v4f*)(sy + e_i);
        #pragma unroll
        for (int j = 0; j < s; ++j) {
          v4f kj = *(const v4f*)(sk + j * (MT * D_N) + e_i);
          const float c = dt * AC[s][j];
          v += c * kj;
        }
        *(v4h*)(hA0 + e_row * HA_S + e_d) = __builtin_convertvector(v, v4h);
      }
      __syncthreads();

      gemm_layer<2, false, true , true >(wave, lane, hA0, nullptr, frag + W1F_OFF,
                                         sb1, hA1, nullptr, oscale);
      __syncthreads();
      gemm_layer<8, true , true , true >(wave, lane, hA1, sW2, nullptr,
                                         sb2, hA0, nullptr, oscale);
      __syncthreads();
      gemm_layer<8, true , true , true >(wave, lane, hA0, sW3, nullptr,
                                         sb3, hA1, nullptr, oscale);
      __syncthreads();
      gemm_layer<8, false, false, false>(wave, lane, hA1, nullptr, frag + W4F_OFF,
                                         sb4, nullptr, sk + s * (MT * D_N), oscale);
      __syncthreads();
    }

    // y_{n+1} = y_n + dt * sum b_i k_i ; store to out[:, t+1, :] (straight-line)
    {
      v4f v = *(const v4f*)(sy + e_i);
      v4f a = BW[0] * (*(const v4f*)(sk + 0 * (MT * D_N) + e_i));
      #pragma unroll
      for (int j = 1; j < 6; ++j)
        a += BW[j] * (*(const v4f*)(sk + j * (MT * D_N) + e_i));
      v += dt * a;
      *(v4f*)(sy + e_i) = v;
      __builtin_nontemporal_store(
          v, (v4f*)(out + (size_t)(row0 + e_row) * (T_N * D_N)
                        + (size_t)(t + 1) * D_N + e_d));
    }
    __syncthreads();
  }
}

extern "C" void kernel_launch(void* const* d_in, const int* in_sizes, int n_in,
                              void* d_out, int out_size, void* d_ws, size_t ws_size,
                              hipStream_t stream) {
  (void)in_sizes; (void)n_in; (void)out_size; (void)ws_size;
  const float* ts  = (const float*)d_in[0];
  const float* y0  = (const float*)d_in[1];
  const float* W1  = (const float*)d_in[2];
  const float* b1  = (const float*)d_in[3];
  const float* W2  = (const float*)d_in[4];
  const float* b2  = (const float*)d_in[5];
  const float* W3  = (const float*)d_in[6];
  const float* b3  = (const float*)d_in[7];
  const float* W4  = (const float*)d_in[8];
  const float* b4  = (const float*)d_in[9];
  const float* osc = (const float*)d_in[10];
  _Float16* frag = (_Float16*)d_ws;
  float* out = (float*)d_out;

  prep_frags<<<(FRAG_TOTAL + 255) / 256, 256, 0, stream>>>(W1, W2, W3, W4, frag);

  constexpr size_t LDS_BYTES =
      (size_t)(W2F_LEN + W3F_LEN + 2 * MT * HA_S) * sizeof(_Float16) +
      (size_t)(MT * D_N + 6 * MT * D_N + 3 * W_N + D_N) * sizeof(float);
  (void)hipFuncSetAttribute(reinterpret_cast<const void*>(tsit5_kernel),
                            hipFuncAttributeMaxDynamicSharedMemorySize, (int)LDS_BYTES);
  tsit5_kernel<<<B_N / MT, 256, LDS_BYTES, stream>>>(ts, y0, b1, b2, b3, b4, osc, frag, out);
}
